// CartesianMoE_49331994362125
// MI455X (gfx1250) — matrix operand order repaired and checked by reading, not compiled, not run on hardware
//
#include <hip/hip_runtime.h>
#include <hip/hip_bf16.h>

typedef __attribute__((ext_vector_type(16))) _Float16 v16h;
typedef __attribute__((ext_vector_type(8)))  _Float16 v8h;
typedef __attribute__((ext_vector_type(8)))  float    v8f;

#define D_MODEL 1024
#define D_FF    2048
#define H2      4096   // 2*D_FF
#define NA      4
#define NB      4
#define NE      16
#define T_TOK   2048
#define CAP     2048   // per-expert slot capacity (worst case: all tokens one expert)
#define M_TILE  32     // tokens per FFN block
#define TILES_PER_E 64 // CAP / M_TILE

static __device__ inline v8f wmma_f16(v16h a, v16h b, v8f c) {
    return __builtin_amdgcn_wmma_f32_16x16x32_f16(
        /*neg_a=*/false, a, /*neg_b=*/false, b,
        /*c_mod=*/(short)0, c, /*reuse_a=*/false, /*reuse_b=*/false);
}

// A-fragment (16x32 f16) from a per-lane row pointer (f16, row-major).
// Lanes 0-15: K = k0+0..7 and k0+16..23 ; lanes 16-31: K = k0+8..15, k0+24..31.
static __device__ inline v16h load_afrag_row(const _Float16* rowp, int k0, int lane) {
    int kh = (lane >> 4) << 3;            // 0 or 8
    const _Float16* p = rowp + k0 + kh;
    v8h lo = *(const v8h*)p;
    v8h hi = *(const v8h*)(p + 16);
    return __builtin_shufflevector(lo, hi, 0,1,2,3,4,5,6,7,8,9,10,11,12,13,14,15);
}

// A-fragment from f16 LDS tile [rows][row_stride], rows offset by rbase0.
static __device__ inline v16h load_afrag_lds(const _Float16* S, int row_stride,
                                             int rbase0, int k0, int lane) {
    int r  = rbase0 + (lane & 15);
    int kh = (lane >> 4) << 3;
    const _Float16* p = S + r * row_stride + k0 + kh;
    v8h lo = *(const v8h*)p;
    v8h hi = *(const v8h*)(p + 16);
    return __builtin_shufflevector(lo, hi, 0,1,2,3,4,5,6,7,8,9,10,11,12,13,14,15);
}

// B-fragment (32x16) from row-major fp32 global matrix, converting to f16.
// Lanes 0-15: col = n0+lane, K = k0+0..7, k0+16..23 ; lanes 16-31: K = k0+8..15, k0+24..31.
static __device__ inline v16h load_bfrag_f32(const float* Bg, int k0, int n0,
                                             int ldb, int lane) {
    int n  = n0 + (lane & 15);
    int kb = k0 + ((lane >> 4) << 3);
    const float* p = Bg + (size_t)kb * ldb + n;
    v16h b;
#pragma unroll
    for (int i = 0; i < 16; ++i) b[i] = (_Float16)p[(size_t)(i + ((i >> 3) << 3)) * ldb];
    return b;
}

// ---------------------------------------------------------------------------
// K0: convert x to f16 in workspace; zero the dummy-row page.
// ---------------------------------------------------------------------------
__global__ __launch_bounds__(256) void xcvt_kernel(
    const float* __restrict__ x, _Float16* __restrict__ xhalf,
    _Float16* __restrict__ zbuf) {
    int i = blockIdx.x * 256 + threadIdx.x;   // grid covers T_TOK*D_MODEL/4
#pragma unroll
    for (int j = 0; j < 4; ++j) {
        int idx = i * 4 + j;
        xhalf[idx] = (_Float16)x[idx];
    }
    if (blockIdx.x == 0 && threadIdx.x < 256) {
#pragma unroll
        for (int j = 0; j < 4; ++j) zbuf[threadIdx.x * 4 + j] = (_Float16)0.f;
    }
}

// ---------------------------------------------------------------------------
// K1: gating. 1 wave per token; 4 waves (128 threads) per block; 512 blocks.
// ---------------------------------------------------------------------------
__global__ __launch_bounds__(128) void gate_kernel(
    const float* __restrict__ x,
    const float* __restrict__ Wga, const float* __restrict__ bga,
    const float* __restrict__ Wgb, const float* __restrict__ bgb,
    int* __restrict__ eIdx, float* __restrict__ partials) {
    const int tid  = threadIdx.x;
    const int lane = tid & 31;
    const int wave = tid >> 5;
    const int t    = blockIdx.x * 4 + wave;

    float aA[NA] = {0.f, 0.f, 0.f, 0.f};
    float aB[NB] = {0.f, 0.f, 0.f, 0.f};
    const float* xr = x + (size_t)t * D_MODEL;
    for (int k = lane; k < D_MODEL; k += 32) {
        float xv = xr[k];
        const float* wa = Wga + k * NA;
        const float* wb = Wgb + k * NB;
#pragma unroll
        for (int j = 0; j < NA; ++j) aA[j] += xv * wa[j];
#pragma unroll
        for (int j = 0; j < NB; ++j) aB[j] += xv * wb[j];
    }
#pragma unroll
    for (int off = 16; off > 0; off >>= 1) {
#pragma unroll
        for (int j = 0; j < NA; ++j) aA[j] += __shfl_xor(aA[j], off, 32);
#pragma unroll
        for (int j = 0; j < NB; ++j) aB[j] += __shfl_xor(aB[j], off, 32);
    }

    __shared__ float sPart[4][16];
    if (lane == 0) {
#pragma unroll
        for (int j = 0; j < NA; ++j) aA[j] += bga[j];
#pragma unroll
        for (int j = 0; j < NB; ++j) aB[j] += bgb[j];
        float mA = aA[0]; int iA = 0;
#pragma unroll
        for (int j = 1; j < NA; ++j) if (aA[j] > mA) { mA = aA[j]; iA = j; }
        float eA[NA], sA = 0.f;
#pragma unroll
        for (int j = 0; j < NA; ++j) { eA[j] = __expf(aA[j] - mA); sA += eA[j]; }
        float mB = aB[0]; int iB = 0;
#pragma unroll
        for (int j = 1; j < NB; ++j) if (aB[j] > mB) { mB = aB[j]; iB = j; }
        float eB[NB], sB = 0.f;
#pragma unroll
        for (int j = 0; j < NB; ++j) { eB[j] = __expf(aB[j] - mB); sB += eB[j]; }

        eIdx[t] = iA * NB + iB;
        float rA = __builtin_amdgcn_rcpf(sA);
        float rB = __builtin_amdgcn_rcpf(sB);
#pragma unroll
        for (int j = 0; j < NA; ++j) sPart[wave][j]      = eA[j] * rA;
#pragma unroll
        for (int j = 0; j < NB; ++j) sPart[wave][4 + j]  = eB[j] * rB;
#pragma unroll
        for (int j = 0; j < NA; ++j) sPart[wave][8 + j]  = (j == iA) ? 1.f : 0.f;
#pragma unroll
        for (int j = 0; j < NB; ++j) sPart[wave][12 + j] = (j == iB) ? 1.f : 0.f;
    }
    __syncthreads();
    if (tid < 16) {
        float s = sPart[0][tid] + sPart[1][tid] + sPart[2][tid] + sPart[3][tid];
        partials[blockIdx.x * 16 + tid] = s;
    }
}

// ---------------------------------------------------------------------------
// K2: finalize (single block). counts, cursor=0, tokmap=-1, aux -> out tail.
// ---------------------------------------------------------------------------
__global__ __launch_bounds__(256) void finalize_kernel(
    const int* __restrict__ eIdx, const float* __restrict__ partials,
    int* __restrict__ counts, int* __restrict__ cursor,
    int* __restrict__ tokmap, float* __restrict__ auxOut) {
    const int tid = threadIdx.x;
    __shared__ int sCnt[NE];
    if (tid < NE) sCnt[tid] = 0;
    __syncthreads();
    for (int t = tid; t < T_TOK; t += 256) atomicAdd(&sCnt[eIdx[t]], 1);
    __syncthreads();
    if (tid < NE) { counts[tid] = sCnt[tid]; cursor[tid] = 0; }
    for (int i = tid; i < NE * CAP; i += 256) tokmap[i] = -1;
    if (tid == 0) {
        float S[16];
#pragma unroll
        for (int j = 0; j < 16; ++j) S[j] = 0.f;
        for (int b = 0; b < 512; ++b)
            for (int j = 0; j < 16; ++j) S[j] += partials[b * 16 + j];
        float auxA = 0.f, auxB = 0.f;
        const float invT = 1.0f / (float)T_TOK;
#pragma unroll
        for (int j = 0; j < NA; ++j) auxA += (S[j] * invT) * (S[8 + j] * invT);
#pragma unroll
        for (int j = 0; j < NB; ++j) auxB += (S[4 + j] * invT) * (S[12 + j] * invT);
        *auxOut = (float)NA * auxA + (float)NB * auxB;
    }
}

// ---------------------------------------------------------------------------
// K3: scatter tokens into per-expert slots (slot order irrelevant to output).
// ---------------------------------------------------------------------------
__global__ __launch_bounds__(256) void scatter_kernel(
    const int* __restrict__ eIdx, int* __restrict__ cursor,
    int* __restrict__ tokmap) {
    int t = blockIdx.x * 256 + threadIdx.x;
    if (t < T_TOK) {
        int e = eIdx[t];
        int pos = atomicAdd(&cursor[e], 1);
        tokmap[e * CAP + pos] = t;
    }
}

// ---------------------------------------------------------------------------
// K4: fused expert FFN for one 32-token tile.
// grid = NE * TILES_PER_E blocks x 256 threads (8 waves = 2 row-tiles x 4 cgs).
// ---------------------------------------------------------------------------
__global__ __launch_bounds__(256) void moe_ffn_kernel(
    const _Float16* __restrict__ xhalf, const _Float16* __restrict__ zbuf,
    const float* __restrict__ W1, const float* __restrict__ b1,
    const float* __restrict__ W2, const float* __restrict__ b2,
    const int* __restrict__ tokmap, const int* __restrict__ counts,
    float* __restrict__ out) {
    const int e    = blockIdx.x >> 6;              // / TILES_PER_E
    const int tile = blockIdx.x & (TILES_PER_E - 1);
    if (tile * M_TILE >= counts[e]) return;        // block-uniform early exit

    const int tid  = threadIdx.x;
    const int lane = tid & 31;
    const int wave = tid >> 5;
    const int rt   = wave >> 2;                    // row-tile 0/1
    const int cg   = wave & 3;                     // column group 0..3
    const int rb0  = rt * 16;                      // row base of this wave

    __shared__ _Float16 sAct[M_TILE * D_FF];       // 128 KB
    __shared__ int sTok[M_TILE];

    if (tid < M_TILE) sTok[tid] = tokmap[e * CAP + tile * M_TILE + tid];
    __syncthreads();

    // Per-lane A-row pointer for GEMM1 (dummy rows -> zero page).
    {
        const int r   = lane & 15;
        const int tok = sTok[rb0 + r];
        const _Float16* aRow = (tok >= 0) ? (xhalf + (size_t)tok * D_MODEL) : zbuf;

        const float* W1e = W1 + (size_t)e * D_MODEL * H2;
        const float* b1e = b1 + e * H2;

        // GEMM1 + SwiGLU: this wave owns 32 paired N-tiles (512 act columns).
        for (int jt = cg * 32; jt < cg * 32 + 32; ++jt) {
            const int n0 = jt * 16;
            v8f ca = {}; v8f cg_acc = {};
            for (int k0 = 0; k0 < D_MODEL; k0 += 32) {
                v16h a  = load_afrag_row(aRow, k0, lane);
                v16h ba = load_bfrag_f32(W1e, k0, n0,        H2, lane);
                v16h bg = load_bfrag_f32(W1e, k0, n0 + D_FF, H2, lane);
                ca     = wmma_f16(a, ba, ca);
                cg_acc = wmma_f16(a, bg, cg_acc);
            }
            const int col   = n0 + (lane & 15);
            const float biA = b1e[col];
            const float biG = b1e[col + D_FF];
            const int rbase = rb0 + ((lane >> 4) << 3);
#pragma unroll
            for (int p = 0; p < 8; ++p) {
                float av = ca[p] + biA;
                float gv = cg_acc[p] + biG;
                float sg = __builtin_amdgcn_rcpf(1.0f + __expf(-gv)); // sigmoid
                float act = av * gv * sg;                             // a*silu(g)
                sAct[(rbase + p) * D_FF + col] = (_Float16)act;
            }
        }
    }
    __syncthreads();

    const float* W2e = W2 + (size_t)e * D_FF * D_MODEL;
    const float* b2e = b2 + e * D_MODEL;

    // GEMM2: this wave owns 16 N-tiles (256 out columns), K = 2048.
    for (int jt = cg * 16; jt < cg * 16 + 16; ++jt) {
        const int n0 = jt * 16;
        v8f c = {};
        for (int k0 = 0; k0 < D_FF; k0 += 32) {
            v16h a = load_afrag_lds(sAct, D_FF, rb0, k0, lane);
            v16h b = load_bfrag_f32(W2e, k0, n0, D_MODEL, lane);
            c = wmma_f16(a, b, c);
        }
        const int col   = n0 + (lane & 15);
        const float bi  = b2e[col];
        const int rbase = rb0 + ((lane >> 4) << 3);
#pragma unroll
        for (int p = 0; p < 8; ++p) {
            int tok = sTok[rbase + p];
            if (tok >= 0) out[(size_t)tok * D_MODEL + col] = c[p] + bi;
        }
    }
}

// ---------------------------------------------------------------------------
extern "C" void kernel_launch(void* const* d_in, const int* in_sizes, int n_in,
                              void* d_out, int out_size, void* d_ws, size_t ws_size,
                              hipStream_t stream) {
    const float* x   = (const float*)d_in[0];
    const float* W1  = (const float*)d_in[1];
    const float* b1  = (const float*)d_in[2];
    const float* W2  = (const float*)d_in[3];
    const float* b2  = (const float*)d_in[4];
    const float* Wga = (const float*)d_in[5];
    const float* bga = (const float*)d_in[6];
    const float* Wgb = (const float*)d_in[7];
    const float* bgb = (const float*)d_in[8];
    float* out = (float*)d_out;

    // Workspace layout
    char* ws = (char*)d_ws;
    int*       eIdx     = (int*)(ws);                   // 2048 ints
    int*       counts   = (int*)(ws + 8192);            // 16 ints
    int*       cursor   = (int*)(ws + 8320);            // 16 ints
    int*       tokmap   = (int*)(ws + 8448);            // 16*2048 ints -> 139520
    float*     partials = (float*)(ws + 139520);        // 512*16 f32   -> 172288
    _Float16*  zbuf     = (_Float16*)(ws + 172288);     // 1024 f16     -> 174336
    _Float16*  xhalf    = (_Float16*)(ws + 176128);     // 2048*1024 f16 (4 MB)

    float* auxOut = out + (size_t)T_TOK * D_MODEL;      // d_out[2097152]

    xcvt_kernel<<<(T_TOK * D_MODEL) / 1024, 256, 0, stream>>>(x, xhalf, zbuf);
    gate_kernel<<<T_TOK / 4, 128, 0, stream>>>(x, Wga, bga, Wgb, bgb,
                                               eIdx, partials);
    finalize_kernel<<<1, 256, 0, stream>>>(eIdx, partials, counts, cursor,
                                           tokmap, auxOut);
    scatter_kernel<<<T_TOK / 256, 256, 0, stream>>>(eIdx, cursor, tokmap);
    moe_ffn_kernel<<<NE * TILES_PER_E, 256, 0, stream>>>(xhalf, zbuf,
                                                         W1, b1, W2, b2,
                                                         tokmap, counts, out);
}
